// EdgeGroupImportanceScorer_34256659153224
// MI455X (gfx1250) — matrix operand order repaired and checked by reading, not compiled, not run on hardware
//
#include <hip/hip_runtime.h>
#include <hip/hip_bf16.h>
#include <stdint.h>

#define E_EDGES   800000
#define G_GROUPS  1000000
#define EDGE_DIM  128
#define NUM_NODES 100000

typedef __attribute__((ext_vector_type(16))) __bf16 v16bf;
typedef __attribute__((ext_vector_type(8)))  float  v8f;

union Frag { v16bf bf; uint4 q[2]; };

// Native bf16 convert (RNE) via clang __bf16; on gfx1250 this selects the hw cvt.
__device__ __forceinline__ unsigned short f2bf(float f) {
  return __builtin_bit_cast(unsigned short, (__bf16)f);
}
__device__ __forceinline__ unsigned pk2bf(float x, float y) {
  return (unsigned)f2bf(x) | ((unsigned)f2bf(y) << 16);
}
__device__ __forceinline__ uint4 pack8(const float* __restrict__ p) {
  float4 a = *(const float4*)p;
  float4 b = *(const float4*)(p + 4);
  uint4 r;
  r.x = pk2bf(a.x, a.y);
  r.y = pk2bf(a.z, a.w);
  r.z = pk2bf(b.x, b.y);
  r.w = pk2bf(b.z, b.w);
  return r;
}

// CDNA5 async copy: global -> LDS, 16B per lane, tracked by ASYNCcnt.
__device__ __forceinline__ void async_cp16(void* lds_dst, const void* gsrc) {
  unsigned lo = (unsigned)(unsigned long long)lds_dst;  // LDS offset = low 32 bits
  asm volatile("global_load_async_to_lds_b128 %0, %1, off"
               :: "v"(lo), "v"(gsrc) : "memory");
}
#define WAIT_ASYNC0() asm volatile("s_wait_asynccnt 0x0" ::: "memory")

// ---------------- prep kernels ----------------

__global__ void egis_zero(float* __restrict__ p, int n) {
  int i = blockIdx.x * blockDim.x + threadIdx.x;
  if (i < n) p[i] = 0.f;
}

__global__ void egis_deg(const long long* __restrict__ ei, float* __restrict__ deg) {
  int i = blockIdx.x * blockDim.x + threadIdx.x;
  if (i < 2 * E_EDGES) atomicAdd(&deg[(int)ei[i]], 1.0f);
}

// W1 (256x256 f32, K-major) -> W1T (256x256 bf16, N-major)
// W2 (256x8  f32, K-major) -> W2T (16x256 bf16, N-major, rows 8..15 zero)
__global__ void egis_prep(const float* __restrict__ W1, const float* __restrict__ W2,
                          unsigned short* __restrict__ W1T, unsigned short* __restrict__ W2T) {
  int t = blockIdx.x * blockDim.x + threadIdx.x;
  if (t < 256 * 256) {
    int n = t >> 8, k = t & 255;
    W1T[t] = f2bf(W1[k * 256 + n]);
  } else if (t < 256 * 256 + 16 * 256) {
    int u = t - 256 * 256;
    int n = u >> 8, k = u & 255;
    W2T[u] = (n < 8) ? f2bf(W2[k * 8 + n]) : (unsigned short)0;
  }
}

// ---------------- main fused kernel ----------------
// block = 128 threads (4 wave32), 64 groups per block, 16 groups per wave.
// Dynamic LDS (96KB): [0,32K) bT0 | [32K,64K) bT1 | [64K,96K) hT ; compT aliases bT0.

__global__ __launch_bounds__(128)
void egis_main(const float* __restrict__ emb,
               const long long* __restrict__ eg,
               const long long* __restrict__ ei,
               const unsigned short* __restrict__ W1T,
               const unsigned short* __restrict__ W2T,
               const float* __restrict__ b1,  const float* __restrict__ b2,
               const float* __restrict__ Ws1, const float* __restrict__ bs1,
               const float* __restrict__ Ws2, const float* __restrict__ bs2,
               const float* __restrict__ Wc1, const float* __restrict__ bc1,
               const float* __restrict__ Wc2, const float* __restrict__ bc2,
               const float* __restrict__ deg,
               float* __restrict__ out)
{
  extern __shared__ unsigned char smem[];
  unsigned short* hT    = (unsigned short*)(smem + 65536);  // [4][16][256] bf16 hidden
  float*          compT = (float*)smem;                     // [4][16][8] f32 (aliases bT0)

  const int tid  = threadIdx.x;
  const int wave = tid >> 5;
  const int lane = tid & 31;
  const int mrow = lane & 15;   // matrix row (A) / col (B,C) this lane owns
  const int hi   = lane >> 4;   // half-wave select in K striping
  const long long gbase = (long long)blockIdx.x * 64 + wave * 16;

  // ---- kick off async staging of W1T chunk 0 into bT0 (32KB, 16B/lane x16) ----
  {
    const char* src = (const char*)W1T;           // chunk 0 byte base
    char*       dst = (char*)smem;                // bT0
#pragma unroll
    for (int i = 0; i < 16; ++i) {
      int off = (tid + i * 128) * 16;
      async_cp16(dst + off, src + off);
    }
  }

  // ---- gather A tile (16 groups x 256 K, bf16) straight into WMMA A-fragment regs ----
  // A 16-bit layout: lane<16 : row=lane,    K = k0+{0..7}, k0+{16..23}
  //                  lane>=16: row=lane-16, K = k0+{8..15}, k0+{24..31}
  const long long ga = gbase + mrow;
  const long long e0 = eg[2 * ga], e1 = eg[2 * ga + 1];
  const float* r0 = emb + e0 * EDGE_DIM;
  const float* r1 = emb + e1 * EDGE_DIM;
  Frag afr[8];
#pragma unroll
  for (int ks = 0; ks < 8; ++ks) {
    int ka = ks * 32 + (hi ? 8 : 0);
    int kb = ka + 16;
    const float* pa = (ka < 128) ? (r0 + ka) : (r1 + ka - 128);
    const float* pb = (kb < 128) ? (r0 + kb) : (r1 + kb - 128);
    afr[ks].q[0] = pack8(pa);
    afr[ks].q[1] = pack8(pb);
  }

  WAIT_ASYNC0();
  __syncthreads();  // chunk 0 resident block-wide

  // ---- stage 1: (16x256) @ W1 (256x256), four 64-wide N chunks, double-buffered ----
  for (int c = 0; c < 4; ++c) {
    if (c < 3) {  // async-prefetch next chunk into the other buffer
      const char* src = (const char*)(W1T + (c + 1) * 64 * 256);
      char*       dst = (char*)smem + (((c + 1) & 1) << 15);
#pragma unroll
      for (int i = 0; i < 16; ++i) {
        int off = (tid + i * 128) * 16;
        async_cp16(dst + off, src + off);
      }
    }
    const unsigned short* buf = (const unsigned short*)((const char*)smem + ((c & 1) << 15));
#pragma unroll
    for (int nt = 0; nt < 4; ++nt) {
      v8f acc = {0.f, 0.f, 0.f, 0.f, 0.f, 0.f, 0.f, 0.f};
      const unsigned short* brow = buf + (nt * 16 + mrow) * 256; // this lane's B column
#pragma unroll
      for (int ks = 0; ks < 8; ++ks) {
        // B 16-bit layout: lane<16 holds K=k0+{0..7},k0+{16..23} of col=lane, lane>=16 K=k0+{8..15},k0+{24..31}
        Frag bfr;
        const uint4* bp = (const uint4*)(brow + ks * 32 + hi * 8);
        bfr.q[0] = bp[0];
        bfr.q[1] = bp[2];
        acc = __builtin_amdgcn_wmma_f32_16x16x32_bf16(false, afr[ks].bf, false, bfr.bf,
                                                      (short)0, acc, false, false);
      }
      // bias + relu; transpose C (col-per-lane) -> hidden tile (row-major bf16) via LDS
      int ng = c * 64 + nt * 16 + mrow;
      float bn = b1[ng];
      unsigned short* hb = hT + wave * 4096 + ng;
#pragma unroll
      for (int r = 0; r < 8; ++r) {
        int m = r + 8 * hi;
        hb[m * 256] = f2bf(fmaxf(acc[r] + bn, 0.f));
      }
    }
    if (c < 3) {
      WAIT_ASYNC0();     // my prefetch landed
      __syncthreads();   // everyone's landed + everyone done reading current buffer
    }
  }
  __syncthreads();  // all waves done with bT buffers; compT may alias bT0

  // ---- stage 2: hidden (16x256) @ W2T (256x8, padded to 16 cols) ----
  v8f acc2 = {0.f, 0.f, 0.f, 0.f, 0.f, 0.f, 0.f, 0.f};
  const unsigned short* hrow  = hT + wave * 4096 + mrow * 256;
  const unsigned short* w2row = W2T + mrow * 256;
#pragma unroll
  for (int ks = 0; ks < 8; ++ks) {
    Frag af2, bf2;
    int ka = ks * 32 + (hi ? 8 : 0);
    af2.q[0] = *(const uint4*)(hrow + ka);
    af2.q[1] = *(const uint4*)(hrow + ka + 16);
    const uint4* bp = (const uint4*)(w2row + ks * 32 + hi * 8);
    bf2.q[0] = bp[0];
    bf2.q[1] = bp[2];
    acc2 = __builtin_amdgcn_wmma_f32_16x16x32_bf16(false, af2.bf, false, bf2.bf,
                                                   (short)0, acc2, false, false);
  }
  if (mrow < 8) {                 // only first 8 output cols are real
    float bn = b2[mrow];
#pragma unroll
    for (int r = 0; r < 8; ++r) {
      int m = r + 8 * hi;
      compT[(wave * 16 + m) * 8 + mrow] = fmaxf(acc2[r] + bn, 0.f);
    }
  }
  __syncthreads();

  // ---- tail: degree MLP + 9->128->1 head, 2 lanes per group ----
  const long long gi = gbase + mrow;
  const long long f0 = eg[2 * gi], f1 = eg[2 * gi + 1];
  const float d_i0 = deg[ei[f0]];
  const float d_i1 = deg[ei[E_EDGES + f0]];
  const float d_j0 = deg[ei[f1]];
  const float d_j1 = deg[ei[E_EDGES + f1]];

  float sp = 0.f;
#pragma unroll 4
  for (int j = 0; j < 32; ++j) {
    int ju = hi * 32 + j;
    float h = d_i0 * Ws1[ju] + d_i1 * Ws1[64 + ju] + d_j0 * Ws1[128 + ju] +
              d_j1 * Ws1[192 + ju] + bs1[ju];
    sp += fmaxf(h, 0.f) * Ws2[ju];
  }
  float ss = sp + __shfl_xor(sp, 16, 32) + bs2[0];

  float a[9];
#pragma unroll
  for (int i = 0; i < 8; ++i) a[i] = compT[(wave * 16 + mrow) * 8 + i];
  a[8] = ss;

  float accf = 0.f;
#pragma unroll 2
  for (int j = 0; j < 64; ++j) {
    int ju = hi * 64 + j;
    float h = bc1[ju];
#pragma unroll
    for (int i = 0; i < 9; ++i) h += a[i] * Wc1[i * 128 + ju];
    accf += fmaxf(h, 0.f) * Wc2[ju];
  }
  float tot = accf + __shfl_xor(accf, 16, 32) + bc2[0];
  if (hi == 0) out[gi] = 1.f / (1.f + expf(-tot));
}

// ---------------- launch ----------------

extern "C" void kernel_launch(void* const* d_in, const int* in_sizes, int n_in,
                              void* d_out, int out_size, void* d_ws, size_t ws_size,
                              hipStream_t stream) {
  const float*     emb = (const float*)d_in[0];
  const long long* eg  = (const long long*)d_in[1];
  const long long* ei  = (const long long*)d_in[2];
  const float* W1  = (const float*)d_in[3];
  const float* b1  = (const float*)d_in[4];
  const float* W2  = (const float*)d_in[5];
  const float* b2  = (const float*)d_in[6];
  const float* Ws1 = (const float*)d_in[7];
  const float* bs1 = (const float*)d_in[8];
  const float* Ws2 = (const float*)d_in[9];
  const float* bs2 = (const float*)d_in[10];
  const float* Wc1 = (const float*)d_in[11];
  const float* bc1 = (const float*)d_in[12];
  const float* Wc2 = (const float*)d_in[13];
  const float* bc2 = (const float*)d_in[14];
  float* out = (float*)d_out;

  char* ws = (char*)d_ws;
  float*          deg = (float*)ws;                              // 400,000 B
  unsigned short* W1T = (unsigned short*)(ws + 401408);          // 131,072 B
  unsigned short* W2T = (unsigned short*)(ws + 401408 + 131072); //   8,192 B

  egis_zero<<<(NUM_NODES + 255) / 256, 256, 0, stream>>>(deg, NUM_NODES);
  egis_deg <<<(2 * E_EDGES + 255) / 256, 256, 0, stream>>>(ei, deg);
  egis_prep<<<(256 * 256 + 16 * 256 + 255) / 256, 256, 0, stream>>>(W1, W2, W1T, W2T);
  egis_main<<<G_GROUPS / 64, 128, 98304, stream>>>(emb, eg, ei, W1T, W2T, b1, b2,
                                                   Ws1, bs1, Ws2, bs2, Wc1, bc1, Wc2, bc2,
                                                   deg, out);
}
